// Model_46660524703814
// MI455X (gfx1250) — matrix operand order repaired and checked by reading, not compiled, not run on hardware
//
#include <hip/hip_runtime.h>
#include <cstdint>

// ---------------- model constants ----------------
#define BB      4
#define LL      128
#define NV      24
#define PREDL   64
#define PTOT    (BB*LL*NV)      // 12288
#define QTOT    (BB*PREDL*NV)   // 6144
#define LBATCH  (LL*NV)         // 3072 points per batch
#define QBATCH  (PREDL*NV)      // 1536 queries per batch
#define DM      128
#define HIDN    256
#define KNN     16
#define DCOORD  32
#define LN_EPS  1e-5f
#define MPTS    4               // points per block in WMMA kernels (M = 64)
#define MROWS   (MPTS*KNN)      // 64

typedef __attribute__((ext_vector_type(16))) _Float16 v16h;
typedef __attribute__((ext_vector_type(8)))  float    v8f;

#define Z8 {0.f,0.f,0.f,0.f,0.f,0.f,0.f,0.f}

// ---------------- WMMA fragment helpers (gfx1250 wave32 layouts) ----------------
// A fragment 16x32 f16: lane r=lane&15 is row M=r; halves 0..7 = K[hi*8 .. hi*8+7],
// halves 8..15 = K[16+hi*8 .. 16+hi*8+7] where hi = lane>>4. Row-major source.
__device__ __forceinline__ v16h frag_a(const _Float16* base, int stride, int lane) {
  const _Float16* p0 = base + (size_t)(lane & 15) * stride + ((lane >> 4) << 3);
  v16h f;
  *((uint4*)&f)     = *(const uint4*)(p0);
  *((uint4*)&f + 1) = *(const uint4*)(p0 + 16);
  return f;
}
// B fragment 32x16 f16: lane n=lane&15 is column N=n; halves 0..7 = K[hi*8 .. hi*8+7],
// halves 8..15 = K[16+hi*8 .. 16+hi*8+7]. Source is W^T stored [N][K] row-major.
__device__ __forceinline__ v16h frag_b(const _Float16* base, int stride, int lane) {
  const _Float16* p0 = base + (size_t)(lane & 15) * stride + ((lane >> 4) << 3);
  v16h f;
  *((uint4*)&f)     = *(const uint4*)(p0);
  *((uint4*)&f + 1) = *(const uint4*)(p0 + 16);
  return f;
}
__device__ __forceinline__ v8f wmma16(v16h a, v16h b, v8f c) {
  return __builtin_amdgcn_wmma_f32_16x16x32_f16(false, a, false, b, (short)0, c,
                                                false, false);
}
// C/D tile scatter: lane holds col n = lane&15, rows m0 + e (m0 includes hi*8).
__device__ __forceinline__ void store_relu_f16p(v8f acc, _Float16* dst, int stride,
                                                int m0, int n, float bias) {
  for (int e = 0; e < 8; ++e) {
    float z = acc[e] + bias;
    dst[(size_t)(m0 + e) * stride + n] = (_Float16)(z > 0.f ? z : 0.f);
  }
}
__device__ __forceinline__ void store_f32p(v8f acc, float* dst, int stride,
                                           int m0, int n, float bias) {
  for (int e = 0; e < 8; ++e) dst[(size_t)(m0 + e) * stride + n] = acc[e] + bias;
}

// ---------------- async gather: rows of 128 f32 from global -> LDS ----------------
__device__ __forceinline__ void async_gather_rows128(const float* __restrict__ src,
                                                     const int* __restrict__ idxN,
                                                     float (*dst)[DM], int rows, int tid) {
  for (int c = tid; c < rows * 32; c += 256) {     // 16B chunks
    int row = c >> 5;
    int off = (c & 31) << 2;                       // float offset
    int j = idxN[row];
    uint32_t laddr = (uint32_t)(uintptr_t)(&dst[row][off]);
    const float* g = src + (size_t)j * DM + off;
    asm volatile("global_load_async_to_lds_b128 %0, %1, off"
                 :: "v"(laddr), "v"(g) : "memory");
  }
  asm volatile("s_wait_asynccnt 0x0" ::: "memory");
}

// ---------------- weight prep: f32 [K][N] -> f16 transposed [N][K] ----------------
__global__ __launch_bounds__(256) void transpose_f16_kernel(
    const float* __restrict__ W, _Float16* __restrict__ Wt, int K, int N) {
  int i = blockIdx.x * 256 + threadIdx.x;
  if (i < K * N) {
    int k = i / N, n = i % N;
    Wt[(size_t)n * K + k] = (_Float16)W[(size_t)k * N + n];
  }
}

// ---------------- embeddings ----------------
__device__ __forceinline__ void time_mlp(float tval,
    const float* W1, const float* b1, const float* W2, const float* b2,
    float* out16) {
  float h1[32];
  for (int j = 0; j < 32; ++j) { float z = tval * W1[j] + b1[j]; h1[j] = z > 0.f ? z : 0.f; }
  for (int o = 0; o < 16; ++o) {
    float a = b2[o];
    for (int j = 0; j < 32; ++j) a += h1[j] * W2[j * 16 + o];
    out16[o] = a;
  }
}

__global__ __launch_bounds__(256) void embed_points_kernel(
    const float* __restrict__ x, const float* __restrict__ x_mark,
    const float* __restrict__ chan_emb,
    const float* teW1, const float* teb1, const float* teW2, const float* teb2,
    const float* veW1, const float* veb1, const float* veW2, const float* veb2,
    float* __restrict__ p_out, float* __restrict__ t_out, float* __restrict__ h_out) {
  int idx = blockIdx.x * 256 + threadIdx.x;
  if (idx >= PTOT) return;
  int v  = idx % NV;
  int bl = idx / NV;                 // b*L + l
  float tval = x_mark[bl];
  float val  = x[idx];
  for (int c = 0; c < 16; ++c) p_out[(size_t)idx * DCOORD + c] = chan_emb[v * 16 + c];
  float et[16];
  time_mlp(tval, teW1, teb1, teW2, teb2, et);
  for (int o = 0; o < 16; ++o) p_out[(size_t)idx * DCOORD + 16 + o] = et[o];
  t_out[idx] = tval;
  float acc[DM];
  for (int o = 0; o < DM; ++o) acc[o] = veb2[o];
  for (int j = 0; j < 256; ++j) {
    float z = val * veW1[j] + veb1[j];
    z = z > 0.f ? z : 0.f;
    for (int o = 0; o < DM; ++o) acc[o] += z * veW2[j * DM + o];
  }
  for (int o = 0; o < DM; ++o) h_out[(size_t)idx * DM + o] = acc[o];
}

__global__ __launch_bounds__(256) void embed_queries_kernel(
    const float* __restrict__ y_mark, const float* __restrict__ chan_emb,
    const float* teW1, const float* teb1, const float* teW2, const float* teb2,
    float* __restrict__ pq_out) {
  int idx = blockIdx.x * 256 + threadIdx.x;
  if (idx >= QTOT) return;
  int v  = idx % NV;
  int bt = idx / NV;                 // b*PRED_L + tp
  float tval = y_mark[bt];
  for (int c = 0; c < 16; ++c) pq_out[(size_t)idx * DCOORD + c] = chan_emb[v * 16 + c];
  float et[16];
  time_mlp(tval, teW1, teb1, teW2, teb2, et);
  for (int o = 0; o < 16; ++o) pq_out[(size_t)idx * DCOORD + 16 + o] = et[o];
}

// ---------------- kNN (fp32 distances, top-16 by insertion) ----------------
__global__ __launch_bounds__(256) void knn_kernel(
    const float* __restrict__ PQ, const float* __restrict__ PH,
    int Qb, int Lb, int* __restrict__ out_idx) {
  __shared__ float s_q[16][DCOORD];
  __shared__ float s_d[16][16];
  __shared__ float s_top_d[16][16];
  __shared__ int   s_top_i[16][16];
  const int tid = threadIdx.x;
  const int batch = blockIdx.y;
  const int q0 = blockIdx.x * 16;
  const float* pq = PQ + (size_t)batch * Qb * DCOORD;
  const float* ph = PH + (size_t)batch * Lb * DCOORD;
  for (int e = tid; e < 16 * DCOORD; e += 256)
    s_q[e >> 5][e & 31] = pq[(size_t)(q0 + (e >> 5)) * DCOORD + (e & 31)];
  { int qq = tid >> 4, kk = tid & 15; s_top_d[qq][kk] = __builtin_inff(); s_top_i[qq][kk] = 0; }
  __syncthreads();
  const int q = tid >> 4, c = tid & 15;
  for (int cb = 0; cb < Lb; cb += 16) {
    const float* pr = ph + (size_t)(cb + c) * DCOORD;
    float d = 0.f;
    for (int k = 0; k < DCOORD; ++k) { float df = s_q[q][k] - pr[k]; d += df * df; }
    s_d[q][c] = d;
    __syncthreads();
    if (tid < 16) {
      for (int cc = 0; cc < 16; ++cc) {
        float dv = s_d[tid][cc];
        if (dv < s_top_d[tid][15]) {
          int gi = batch * Lb + cb + cc;
          int kk = 15;
          while (kk > 0 && s_top_d[tid][kk - 1] > dv) {
            s_top_d[tid][kk] = s_top_d[tid][kk - 1];
            s_top_i[tid][kk] = s_top_i[tid][kk - 1];
            --kk;
          }
          s_top_d[tid][kk] = dv; s_top_i[tid][kk] = gi;
        }
      }
    }
    __syncthreads();
  }
  if (tid < 16) {
    int row = batch * Qb + q0 + tid;
    for (int k = 0; k < KNN; ++k) out_idx[(size_t)row * KNN + k] = s_top_i[tid][k];
  }
}

// ---------------- propagation layer (4 points per block, M=64, 8 waves) --------
__global__ __launch_bounds__(256) void prop_layer_kernel(
    const float* __restrict__ h_in, float* __restrict__ h_out,
    const float* __restrict__ p, const float* __restrict__ t,
    const int* __restrict__ nidx,
    const _Float16* __restrict__ relW1t, const float* __restrict__ relb1,
    const float* __restrict__ relW2,     const float* __restrict__ relb2,
    const _Float16* __restrict__ msgW1t, const float* __restrict__ msgb1,
    const _Float16* __restrict__ msgW2t, const float* __restrict__ msgb2,
    const float* __restrict__ updW1, const float* __restrict__ updb1,
    const float* __restrict__ updW2, const float* __restrict__ updb2,
    const float* __restrict__ ln_g, const float* __restrict__ ln_b) {
  __shared__ __align__(16) _Float16 s_rel[MROWS][288];
  __shared__ __align__(16) _Float16 s_msgf[MROWS][160];
  __shared__ __align__(16) _Float16 s_hid[MROWS][HIDN];
  __shared__ __align__(16) float s_f32buf[MROWS][DM];   // gather stage, then msg out
  __shared__ float s_pi[MPTS][DCOORD];
  __shared__ float s_hi[MPTS][DM];
  __shared__ float s_ti[MPTS];
  __shared__ float s_tj[MROWS];
  __shared__ int   s_nb[MROWS];
  __shared__ float s_scores[MROWS], s_w[MROWS];
  __shared__ float s_agg[MPTS][DM];
  __shared__ float s_hid2[MPTS][HIDN];
  __shared__ float s_red[MPTS][2];

  const int tid = threadIdx.x;
  const int lane = tid & 31, wv = tid >> 5;
  const int i0 = blockIdx.x * MPTS;
  const int* idx64 = nidx + (size_t)i0 * KNN;

  async_gather_rows128(h_in, idx64, s_f32buf, MROWS, tid);   // h_j rows (async)
  for (int e = tid; e < MPTS * DCOORD; e += 256)
    s_pi[e >> 5][e & 31] = p[(size_t)(i0 + (e >> 5)) * DCOORD + (e & 31)];
  for (int e = tid; e < MPTS * DM; e += 256)
    s_hi[e >> 7][e & 127] = h_in[(size_t)(i0 + (e >> 7)) * DM + (e & 127)];
  if (tid < MROWS) { int j = idx64[tid]; s_nb[tid] = j; s_tj[tid] = t[j]; }
  if (tid < MPTS)  s_ti[tid] = t[i0 + tid];
  __syncthreads();

  // rel features [64][288] = [p_rel | h_i | h_j], f16 (row m: pt=m>>4, nb=m&15)
  for (int e = tid; e < MROWS * 288; e += 256) {
    int m = e / 288, cc = e % 288, pt = m >> 4;
    float vf;
    if (cc < 32)       vf = s_pi[pt][cc] - p[(size_t)s_nb[m] * DCOORD + cc];
    else if (cc < 160) vf = s_hi[pt][cc - 32];
    else               vf = s_f32buf[m][cc - 160];
    s_rel[m][cc] = (_Float16)vf;
  }
  // msg features [64][160] = [h_j | p_rel], f16
  for (int e = tid; e < MROWS * 160; e += 256) {
    int m = e / 160, cc = e % 160, pt = m >> 4;
    float vf;
    if (cc < 128) vf = s_f32buf[m][cc];
    else          vf = s_pi[pt][cc - 128] - p[(size_t)s_nb[m] * DCOORD + (cc - 128)];
    s_msgf[m][cc] = (_Float16)vf;
  }
  __syncthreads();

  // rel hidden: [64x288] @ [288x256], relu -> s_hid (B frag reused by 4 M-tiles)
  for (int nt = wv; nt < 16; nt += 8) {
    v8f a0 = Z8, a1 = Z8, a2 = Z8, a3 = Z8;
    for (int kt = 0; kt < 9; ++kt) {
      v16h b = frag_b(relW1t + (size_t)(nt * 16) * 288 + kt * 32, 288, lane);
      a0 = wmma16(frag_a(&s_rel[ 0][kt * 32], 288, lane), b, a0);
      a1 = wmma16(frag_a(&s_rel[16][kt * 32], 288, lane), b, a1);
      a2 = wmma16(frag_a(&s_rel[32][kt * 32], 288, lane), b, a2);
      a3 = wmma16(frag_a(&s_rel[48][kt * 32], 288, lane), b, a3);
    }
    int n = (nt << 4) + (lane & 15), mb = (lane >> 4) << 3;
    float bias = relb1[n];
    store_relu_f16p(a0, &s_hid[0][0], HIDN,      mb, n, bias);
    store_relu_f16p(a1, &s_hid[0][0], HIDN, 16 + mb, n, bias);
    store_relu_f16p(a2, &s_hid[0][0], HIDN, 32 + mb, n, bias);
    store_relu_f16p(a3, &s_hid[0][0], HIDN, 48 + mb, n, bias);
  }
  __syncthreads();

  // rel layer2 (256 -> 1) + causal mask
  if (tid < MROWS) {
    int pt = tid >> 4;
    float s = relb2[0];
    for (int j = 0; j < HIDN; ++j) s += (float)s_hid[tid][j] * relW2[j];
    if (s_tj[tid] > s_ti[pt]) s = -__builtin_inff();
    s_scores[tid] = s;
  }
  __syncthreads();
  if (tid < MPTS) {
    int b0 = tid * KNN;
    float mx = s_scores[b0];
    for (int m = 1; m < KNN; ++m) mx = fmaxf(mx, s_scores[b0 + m]);
    float sum = 0.f;
    for (int m = 0; m < KNN; ++m) { float ez = __expf(s_scores[b0 + m] - mx); s_w[b0 + m] = ez; sum += ez; }
    float inv = 1.f / sum;
    for (int m = 0; m < KNN; ++m) s_w[b0 + m] *= inv;
  }
  __syncthreads();

  // msg hidden: [64x160] @ [160x256], relu -> s_hid
  for (int nt = wv; nt < 16; nt += 8) {
    v8f a0 = Z8, a1 = Z8, a2 = Z8, a3 = Z8;
    for (int kt = 0; kt < 5; ++kt) {
      v16h b = frag_b(msgW1t + (size_t)(nt * 16) * 160 + kt * 32, 160, lane);
      a0 = wmma16(frag_a(&s_msgf[ 0][kt * 32], 160, lane), b, a0);
      a1 = wmma16(frag_a(&s_msgf[16][kt * 32], 160, lane), b, a1);
      a2 = wmma16(frag_a(&s_msgf[32][kt * 32], 160, lane), b, a2);
      a3 = wmma16(frag_a(&s_msgf[48][kt * 32], 160, lane), b, a3);
    }
    int n = (nt << 4) + (lane & 15), mb = (lane >> 4) << 3;
    float bias = msgb1[n];
    store_relu_f16p(a0, &s_hid[0][0], HIDN,      mb, n, bias);
    store_relu_f16p(a1, &s_hid[0][0], HIDN, 16 + mb, n, bias);
    store_relu_f16p(a2, &s_hid[0][0], HIDN, 32 + mb, n, bias);
    store_relu_f16p(a3, &s_hid[0][0], HIDN, 48 + mb, n, bias);
  }
  __syncthreads();

  // msg out: [64x256] @ [256x128] -> s_f32buf (one N-tile per wave)
  {
    v8f a0 = Z8, a1 = Z8, a2 = Z8, a3 = Z8;
    for (int kt = 0; kt < 8; ++kt) {
      v16h b = frag_b(msgW2t + (size_t)(wv * 16) * HIDN + kt * 32, HIDN, lane);
      a0 = wmma16(frag_a(&s_hid[ 0][kt * 32], HIDN, lane), b, a0);
      a1 = wmma16(frag_a(&s_hid[16][kt * 32], HIDN, lane), b, a1);
      a2 = wmma16(frag_a(&s_hid[32][kt * 32], HIDN, lane), b, a2);
      a3 = wmma16(frag_a(&s_hid[48][kt * 32], HIDN, lane), b, a3);
    }
    int n = (wv << 4) + (lane & 15), mb = (lane >> 4) << 3;
    float bias = msgb2[n];
    store_f32p(a0, &s_f32buf[0][0], DM,      mb, n, bias);
    store_f32p(a1, &s_f32buf[0][0], DM, 16 + mb, n, bias);
    store_f32p(a2, &s_f32buf[0][0], DM, 32 + mb, n, bias);
    store_f32p(a3, &s_f32buf[0][0], DM, 48 + mb, n, bias);
  }
  __syncthreads();

  // weighted aggregation: agg[pt][n] = sum_m w * msg
  for (int e = tid; e < MPTS * DM; e += 256) {
    int pt = e >> 7, n = e & 127;
    float a = 0.f;
    for (int m = 0; m < KNN; ++m) a += s_w[pt * KNN + m] * s_f32buf[pt * KNN + m][n];
    s_agg[pt][n] = a;
  }
  __syncthreads();

  // upd MLP (per-point GEMV in fp32)
  for (int e = tid; e < MPTS * HIDN; e += 256) {
    int pt = e >> 8, j = e & 255;
    float a = updb1[j];
    for (int n = 0; n < DM; ++n) a += s_agg[pt][n] * updW1[n * HIDN + j];
    s_hid2[pt][j] = a > 0.f ? a : 0.f;
  }
  __syncthreads();
  for (int e = tid; e < MPTS * DM; e += 256) {
    int pt = e >> 7, n = e & 127;
    float a = updb2[n];
    for (int j = 0; j < HIDN; ++j) a += s_hid2[pt][j] * updW2[j * DM + n];
    s_agg[pt][n] = s_hi[pt][n] + a;     // hn (residual), reuse s_agg
  }
  __syncthreads();
  if (tid < MPTS) {
    float mu = 0.f;
    for (int n = 0; n < DM; ++n) mu += s_agg[tid][n];
    mu *= (1.f / DM);
    float var = 0.f;
    for (int n = 0; n < DM; ++n) { float d = s_agg[tid][n] - mu; var += d * d; }
    var *= (1.f / DM);
    s_red[tid][0] = mu; s_red[tid][1] = rsqrtf(var + LN_EPS);
  }
  __syncthreads();
  for (int e = tid; e < MPTS * DM; e += 256) {
    int pt = e >> 7, n = e & 127;
    h_out[(size_t)(i0 + pt) * DM + n] =
        (s_agg[pt][n] - s_red[pt][0]) * s_red[pt][1] * ln_g[n] + ln_b[n];
  }
}

// ---------------- query decode (4 queries per block, M=64) ----------------
__global__ __launch_bounds__(256) void query_out_kernel(
    const float* __restrict__ h, const float* __restrict__ p,
    const float* __restrict__ pq, const int* __restrict__ qnidx,
    const _Float16* __restrict__ qkW1t, const float* __restrict__ qkb1,
    const float* __restrict__ qkW2,     const float* __restrict__ qkb2,
    const _Float16* __restrict__ vW1t,  const float* __restrict__ vb1,
    const _Float16* __restrict__ vW2t,  const float* __restrict__ vb2,
    const float* __restrict__ rgW1, const float* __restrict__ rgb1,
    const float* __restrict__ rgW2, const float* __restrict__ rgb2,
    float* __restrict__ out) {
  __shared__ __align__(16) _Float16 s_qf[MROWS][160];
  __shared__ __align__(16) _Float16 s_vf[MROWS][DM];
  __shared__ __align__(16) _Float16 s_hid[MROWS][HIDN];
  __shared__ __align__(16) float s_f32buf[MROWS][DM];   // gather stage, then h_vals
  __shared__ float s_pq[MPTS][DCOORD];
  __shared__ int   s_nb[MROWS];
  __shared__ float s_scores[MROWS], s_w[MROWS];
  __shared__ float s_hq[MPTS][DM];
  __shared__ float s_hid2[MPTS][HIDN];

  const int tid = threadIdx.x;
  const int lane = tid & 31, wv = tid >> 5;
  const int q0 = blockIdx.x * MPTS;
  const int* idx64 = qnidx + (size_t)q0 * KNN;

  async_gather_rows128(h, idx64, s_f32buf, MROWS, tid);
  for (int e = tid; e < MPTS * DCOORD; e += 256)
    s_pq[e >> 5][e & 31] = pq[(size_t)(q0 + (e >> 5)) * DCOORD + (e & 31)];
  if (tid < MROWS) s_nb[tid] = idx64[tid];
  __syncthreads();

  for (int e = tid; e < MROWS * 160; e += 256) {
    int m = e / 160, cc = e % 160, pt = m >> 4;
    float vf;
    if (cc < 32) vf = s_pq[pt][cc] - p[(size_t)s_nb[m] * DCOORD + cc];
    else         vf = s_f32buf[m][cc - 32];
    s_qf[m][cc] = (_Float16)vf;
  }
  for (int e = tid; e < MROWS * DM; e += 256) {
    int m = e >> 7, cc = e & 127;
    s_vf[m][cc] = (_Float16)s_f32buf[m][cc];
  }
  __syncthreads();

  // query_kernel hidden: [64x160] @ [160x256], relu
  for (int nt = wv; nt < 16; nt += 8) {
    v8f a0 = Z8, a1 = Z8, a2 = Z8, a3 = Z8;
    for (int kt = 0; kt < 5; ++kt) {
      v16h b = frag_b(qkW1t + (size_t)(nt * 16) * 160 + kt * 32, 160, lane);
      a0 = wmma16(frag_a(&s_qf[ 0][kt * 32], 160, lane), b, a0);
      a1 = wmma16(frag_a(&s_qf[16][kt * 32], 160, lane), b, a1);
      a2 = wmma16(frag_a(&s_qf[32][kt * 32], 160, lane), b, a2);
      a3 = wmma16(frag_a(&s_qf[48][kt * 32], 160, lane), b, a3);
    }
    int n = (nt << 4) + (lane & 15), mb = (lane >> 4) << 3;
    float bias = qkb1[n];
    store_relu_f16p(a0, &s_hid[0][0], HIDN,      mb, n, bias);
    store_relu_f16p(a1, &s_hid[0][0], HIDN, 16 + mb, n, bias);
    store_relu_f16p(a2, &s_hid[0][0], HIDN, 32 + mb, n, bias);
    store_relu_f16p(a3, &s_hid[0][0], HIDN, 48 + mb, n, bias);
  }
  __syncthreads();
  if (tid < MROWS) {
    float s = qkb2[0];
    for (int j = 0; j < HIDN; ++j) s += (float)s_hid[tid][j] * qkW2[j];
    s_scores[tid] = s;
  }
  __syncthreads();
  if (tid < MPTS) {
    int b0 = tid * KNN;
    float mx = s_scores[b0];
    for (int m = 1; m < KNN; ++m) mx = fmaxf(mx, s_scores[b0 + m]);
    float sum = 0.f;
    for (int m = 0; m < KNN; ++m) { float ez = __expf(s_scores[b0 + m] - mx); s_w[b0 + m] = ez; sum += ez; }
    float inv = 1.f / sum;
    for (int m = 0; m < KNN; ++m) s_w[b0 + m] *= inv;
  }
  __syncthreads();

  // value hidden: [64x128] @ [128x128], relu -> s_hid[:, 0:128]
  {
    v8f a0 = Z8, a1 = Z8, a2 = Z8, a3 = Z8;
    for (int kt = 0; kt < 4; ++kt) {
      v16h b = frag_b(vW1t + (size_t)(wv * 16) * DM + kt * 32, DM, lane);
      a0 = wmma16(frag_a(&s_vf[ 0][kt * 32], DM, lane), b, a0);
      a1 = wmma16(frag_a(&s_vf[16][kt * 32], DM, lane), b, a1);
      a2 = wmma16(frag_a(&s_vf[32][kt * 32], DM, lane), b, a2);
      a3 = wmma16(frag_a(&s_vf[48][kt * 32], DM, lane), b, a3);
    }
    int n = (wv << 4) + (lane & 15), mb = (lane >> 4) << 3;
    float bias = vb1[n];
    store_relu_f16p(a0, &s_hid[0][0], HIDN,      mb, n, bias);
    store_relu_f16p(a1, &s_hid[0][0], HIDN, 16 + mb, n, bias);
    store_relu_f16p(a2, &s_hid[0][0], HIDN, 32 + mb, n, bias);
    store_relu_f16p(a3, &s_hid[0][0], HIDN, 48 + mb, n, bias);
  }
  __syncthreads();
  // value out: [64x128] @ [128x128] -> s_f32buf (f32)
  {
    v8f a0 = Z8, a1 = Z8, a2 = Z8, a3 = Z8;
    for (int kt = 0; kt < 4; ++kt) {
      v16h b = frag_b(vW2t + (size_t)(wv * 16) * DM + kt * 32, DM, lane);
      a0 = wmma16(frag_a(&s_hid[ 0][kt * 32], HIDN, lane), b, a0);
      a1 = wmma16(frag_a(&s_hid[16][kt * 32], HIDN, lane), b, a1);
      a2 = wmma16(frag_a(&s_hid[32][kt * 32], HIDN, lane), b, a2);
      a3 = wmma16(frag_a(&s_hid[48][kt * 32], HIDN, lane), b, a3);
    }
    int n = (wv << 4) + (lane & 15), mb = (lane >> 4) << 3;
    float bias = vb2[n];
    store_f32p(a0, &s_f32buf[0][0], DM,      mb, n, bias);
    store_f32p(a1, &s_f32buf[0][0], DM, 16 + mb, n, bias);
    store_f32p(a2, &s_f32buf[0][0], DM, 32 + mb, n, bias);
    store_f32p(a3, &s_f32buf[0][0], DM, 48 + mb, n, bias);
  }
  __syncthreads();
  for (int e = tid; e < MPTS * DM; e += 256) {
    int pt = e >> 7, n = e & 127;
    float a = 0.f;
    for (int m = 0; m < KNN; ++m) a += s_w[pt * KNN + m] * s_f32buf[pt * KNN + m][n];
    s_hq[pt][n] = a;
  }
  __syncthreads();
  for (int e = tid; e < MPTS * HIDN; e += 256) {
    int pt = e >> 8, j = e & 255;
    float a = rgb1[j];
    for (int n = 0; n < DM; ++n) a += s_hq[pt][n] * rgW1[n * HIDN + j];
    s_hid2[pt][j] = a > 0.f ? a : 0.f;
  }
  __syncthreads();
  if (tid < MPTS) {
    float a = rgb2[0];
    for (int j = 0; j < HIDN; ++j) a += s_hid2[tid][j] * rgW2[j];
    out[q0 + tid] = a;
  }
}

// ---------------- host-side launch ----------------
extern "C" void kernel_launch(void* const* d_in, const int* in_sizes, int n_in,
                              void* d_out, int out_size, void* d_ws, size_t ws_size,
                              hipStream_t stream) {
  (void)in_sizes; (void)n_in; (void)out_size; (void)ws_size;
  // pytree order: top-level dict insertion order {x, x_mark, y_mark, params},
  // params dict keys sorted: chan_emb, layers, query_kernel, regr, time_enc,
  // val_enc, value_mlp; layer dicts sorted: ln_b, ln_g, msg, rel, upd.
  const float* x        = (const float*)d_in[0];
  const float* x_mark   = (const float*)d_in[1];
  const float* y_mark   = (const float*)d_in[2];
  const float* chan_emb = (const float*)d_in[3];
  const float* L_ln_b[2]  = {(const float*)d_in[4],  (const float*)d_in[18]};
  const float* L_ln_g[2]  = {(const float*)d_in[5],  (const float*)d_in[19]};
  const float* L_msgW1[2] = {(const float*)d_in[6],  (const float*)d_in[20]};
  const float* L_msgb1[2] = {(const float*)d_in[7],  (const float*)d_in[21]};
  const float* L_msgW2[2] = {(const float*)d_in[8],  (const float*)d_in[22]};
  const float* L_msgb2[2] = {(const float*)d_in[9],  (const float*)d_in[23]};
  const float* L_relW1[2] = {(const float*)d_in[10], (const float*)d_in[24]};
  const float* L_relb1[2] = {(const float*)d_in[11], (const float*)d_in[25]};
  const float* L_relW2[2] = {(const float*)d_in[12], (const float*)d_in[26]};
  const float* L_relb2[2] = {(const float*)d_in[13], (const float*)d_in[27]};
  const float* L_updW1[2] = {(const float*)d_in[14], (const float*)d_in[28]};
  const float* L_updb1[2] = {(const float*)d_in[15], (const float*)d_in[29]};
  const float* L_updW2[2] = {(const float*)d_in[16], (const float*)d_in[30]};
  const float* L_updb2[2] = {(const float*)d_in[17], (const float*)d_in[31]};
  const float* qkW1 = (const float*)d_in[32];
  const float* qkb1 = (const float*)d_in[33];
  const float* qkW2 = (const float*)d_in[34];
  const float* qkb2 = (const float*)d_in[35];
  const float* rgW1 = (const float*)d_in[36];
  const float* rgb1 = (const float*)d_in[37];
  const float* rgW2 = (const float*)d_in[38];
  const float* rgb2 = (const float*)d_in[39];
  const float* teW1 = (const float*)d_in[40];
  const float* teb1 = (const float*)d_in[41];
  const float* teW2 = (const float*)d_in[42];
  const float* teb2 = (const float*)d_in[43];
  const float* veW1 = (const float*)d_in[44];
  const float* veb1 = (const float*)d_in[45];
  const float* veW2 = (const float*)d_in[46];
  const float* veb2 = (const float*)d_in[47];
  const float* vW1  = (const float*)d_in[48];
  const float* vb1  = (const float*)d_in[49];
  const float* vW2  = (const float*)d_in[50];
  const float* vb2  = (const float*)d_in[51];

  // workspace layout (floats)
  float* ws = (float*)d_ws;
  float* p_pts = ws;                                // P*32
  float* p_q   = p_pts + (size_t)PTOT * DCOORD;     // Q*32
  float* t_pts = p_q   + (size_t)QTOT * DCOORD;     // P
  float* h_a   = t_pts + PTOT;                      // P*128
  float* h_b   = h_a   + (size_t)PTOT * DM;         // P*128
  int*   nidx  = (int*)(h_b + (size_t)PTOT * DM);   // P*16
  int*   qnidx = nidx + (size_t)PTOT * KNN;         // Q*16
  _Float16* wt = (_Float16*)(qnidx + (size_t)QTOT * KNN);
  _Float16* relW1t[2]; _Float16* msgW1t[2]; _Float16* msgW2t[2];
  relW1t[0] = wt;                  // 256*288
  msgW1t[0] = relW1t[0] + 256 * 288;
  msgW2t[0] = msgW1t[0] + 256 * 160;
  relW1t[1] = msgW2t[0] + 128 * 256;
  msgW1t[1] = relW1t[1] + 256 * 288;
  msgW2t[1] = msgW1t[1] + 256 * 160;
  _Float16* qkW1t = msgW2t[1] + 128 * 256;
  _Float16* vW1t  = qkW1t + 256 * 160;
  _Float16* vW2t  = vW1t + 128 * 128;

  auto tr = [&](const float* W, _Float16* Wt, int K, int N) {
    int total = K * N;
    transpose_f16_kernel<<<(total + 255) / 256, 256, 0, stream>>>(W, Wt, K, N);
  };
  for (int l = 0; l < 2; ++l) {
    tr(L_relW1[l], relW1t[l], 288, 256);
    tr(L_msgW1[l], msgW1t[l], 160, 256);
    tr(L_msgW2[l], msgW2t[l], 256, 128);
  }
  tr(qkW1, qkW1t, 160, 256);
  tr(vW1,  vW1t,  128, 128);
  tr(vW2,  vW2t,  128, 128);

  embed_points_kernel<<<PTOT / 256, 256, 0, stream>>>(
      x, x_mark, chan_emb, teW1, teb1, teW2, teb2,
      veW1, veb1, veW2, veb2, p_pts, t_pts, h_a);
  embed_queries_kernel<<<QTOT / 256, 256, 0, stream>>>(
      y_mark, chan_emb, teW1, teb1, teW2, teb2, p_q);

  knn_kernel<<<dim3(LBATCH / 16, BB), 256, 0, stream>>>(p_pts, p_pts, LBATCH, LBATCH, nidx);
  knn_kernel<<<dim3(QBATCH / 16, BB), 256, 0, stream>>>(p_q, p_pts, QBATCH, LBATCH, qnidx);

  const float* hin = h_a;
  float* hout = h_b;
  for (int l = 0; l < 2; ++l) {
    prop_layer_kernel<<<PTOT / MPTS, 256, 0, stream>>>(
        hin, hout, p_pts, t_pts, nidx,
        relW1t[l], L_relb1[l], L_relW2[l], L_relb2[l],
        msgW1t[l], L_msgb1[l], msgW2t[l], L_msgb2[l],
        L_updW1[l], L_updb1[l], L_updW2[l], L_updb2[l],
        L_ln_g[l], L_ln_b[l]);
    const float* tmp = hin; hin = hout; hout = (float*)tmp;
  }

  query_out_kernel<<<QTOT / MPTS, 256, 0, stream>>>(
      hin, p_pts, p_q, qnidx,
      qkW1t, qkb1, qkW2, qkb2,
      vW1t, vb1, vW2t, vb2,
      rgW1, rgb1, rgW2, rgb2,
      (float*)d_out);
}
